// ByteContextAwareRouter_80461917323983
// MI455X (gfx1250) — matrix operand order repaired and checked by reading, not compiled, not run on hardware
//
#include <hip/hip_runtime.h>

typedef __attribute__((ext_vector_type(16))) __bf16 v16bf;
typedef __attribute__((ext_vector_type(8)))  float  v8f;

union FragAB { unsigned u[8]; uint4 q[2]; v16bf v; };
union FragC  { float f[8]; v8f v; };

#define DEV __device__ __forceinline__

constexpr int N_TOK = 8192;       // B*S = 4*2048
constexpr int H     = 1024;
constexpr int H2    = 2048;
constexpr int CTXD  = 64;
constexpr int POSD  = 32;
constexpr int DG    = 1120;       // H + CTX + POS_D
constexpr int HID   = 3072;
constexpr int NE    = 8;
constexpr int CAP   = 2048;
constexpr int NK    = N_TOK * 2;  // 16384

DEV unsigned short f2bf(float f) {  // round-to-nearest-even fp32 -> bf16
  unsigned u = __float_as_uint(f);
  u += 0x7fffu + ((u >> 16) & 1u);
  return (unsigned short)(u >> 16);
}
DEV float gelu_exact(float v) { return 0.5f * v * (1.0f + erff(v * 0.70710678118654752f)); }

// ---------------------------------------------------------------- conversions
__global__ void k_f32_to_bf16(const float* __restrict__ in,
                              unsigned short* __restrict__ out, int n) {
  int i = blockIdx.x * 256 + threadIdx.x;
  if (i < n) out[i] = f2bf(in[i]);
}

// Pack fp32 weight W[Kd x Nd] (row-major) into WMMA bf16 B-fragment tiles.
// Tile (nt,kt): 256 dwords, lane-major; lane l covers N = nt*16 + (l&15),
// dword i holds K = kt*32 + (i>=4?16:0) + (l>=16?8:0) + 2(i&3), +1  (dense 32x16 bf16 B layout).
__global__ void k_pack_b(const float* __restrict__ W, unsigned* __restrict__ out,
                         int Nd, int Ktiles) {
  int t = blockIdx.x * 256 + threadIdx.x;
  int total = (Nd >> 4) * Ktiles * 256;
  if (t >= total) return;
  int r = t & 255, tile = t >> 8;
  int kt = tile % Ktiles, nt = tile / Ktiles;
  int lane = r >> 3, i = r & 7;
  int n  = nt * 16 + (lane & 15);
  int kb = kt * 32 + ((i >> 2) << 4) + ((lane >> 4) << 3) + 2 * (i & 3);
  unsigned short lo = f2bf(W[(size_t)kb * Nd + n]);
  unsigned short hi = f2bf(W[(size_t)(kb + 1) * Nd + n]);
  out[t] = ((unsigned)hi << 16) | lo;
}

// -------------------------------------------------------------- WMMA GEMM
// Wave computes a 32 x (NT*16) output block (MT=2 M-tiles reuse each B frag).
// A: row-major bf16, Kd2 = Kdim/2 dwords/row. A-fragment per ISA 16-bit A
// table: dwords {0..3}+off, {8..11}+off (off = 4 for lanes>=16).
// EPI: 0 = bias+gelu -> bf16, 1 = bias -> f32, 2 = residual add -> f32.
template <int NT, int EPI>
__global__ void k_gemm_wmma(const unsigned* __restrict__ A,
                            const unsigned* __restrict__ Bp,
                            void* __restrict__ outp,
                            const float* __restrict__ extra,
                            int Ngroups, int Nd, int Ktiles, int Kd2) {
  const int lane = threadIdx.x & 31;
  const int gw = blockIdx.x * (blockDim.x >> 5) + (threadIdx.x >> 5);
  const int mg = gw / Ngroups, ng = gw % Ngroups;   // mg: 32-row group
  const int lane15 = lane & 15, laneHi = lane >> 4;

  FragC acc[2][NT];
  for (int m = 0; m < 2; ++m)
    for (int t = 0; t < NT; ++t)
      for (int r = 0; r < 8; ++r) acc[m][t].f[r] = 0.0f;

  const unsigned* abase0 = A + (size_t)(mg * 32 + lane15) * Kd2 + laneHi * 4;
  const unsigned* abase1 = abase0 + (size_t)16 * Kd2;
  const unsigned* bbase  = Bp + (size_t)(ng * NT) * Ktiles * 256 + lane * 8;

  for (int kt = 0; kt < Ktiles; ++kt) {
    FragAB a0, a1;
    a0.q[0] = *(const uint4*)(abase0 + kt * 16);
    a0.q[1] = *(const uint4*)(abase0 + kt * 16 + 8);
    a1.q[0] = *(const uint4*)(abase1 + kt * 16);
    a1.q[1] = *(const uint4*)(abase1 + kt * 16 + 8);
#pragma unroll
    for (int t = 0; t < NT; ++t) {
      FragAB b;
      const unsigned* bp = bbase + ((size_t)t * Ktiles + kt) * 256;
      b.q[0] = *(const uint4*)bp;
      b.q[1] = *(const uint4*)(bp + 4);
      acc[0][t].v = __builtin_amdgcn_wmma_f32_16x16x32_bf16(
          false, a0.v, false, b.v, (short)0, acc[0][t].v, false, false);
      acc[1][t].v = __builtin_amdgcn_wmma_f32_16x16x32_bf16(
          false, a1.v, false, b.v, (short)0, acc[1][t].v, false, false);
    }
  }

#pragma unroll
  for (int m = 0; m < 2; ++m) {
    const int row0 = mg * 32 + m * 16 + laneHi * 8;
#pragma unroll
    for (int t = 0; t < NT; ++t) {
      const int col = (ng * NT + t) * 16 + lane15;
#pragma unroll
      for (int r = 0; r < 8; ++r) {
        const size_t idx = (size_t)(row0 + r) * Nd + col;
        float v = acc[m][t].f[r];
        if (EPI == 0) {
          v = gelu_exact(v + extra[col]);
          ((unsigned short*)outp)[idx] = f2bf(v);
        } else if (EPI == 1) {
          ((float*)outp)[idx] = v + extra[col];
        } else {
          ((float*)outp)[idx] = v + extra[idx];
        }
      }
    }
  }
}

// Dual GEMM: hmid = silu(gi@w1) * (gi@w3), bf16 out. MT=2 x NT=2; B fragments
// shared across the two M-tiles, A fragment shared across u/v products.
__global__ void k_gemm_swiglu(const unsigned* __restrict__ A,
                              const unsigned* __restrict__ B1,
                              const unsigned* __restrict__ B3,
                              unsigned short* __restrict__ out,
                              int Ngroups, int Nd, int Ktiles, int Kd2) {
  const int lane = threadIdx.x & 31;
  const int gw = blockIdx.x * (blockDim.x >> 5) + (threadIdx.x >> 5);
  const int mg = gw / Ngroups, ng = gw % Ngroups;
  const int lane15 = lane & 15, laneHi = lane >> 4;

  FragC au[2][2], av[2][2];
  for (int m = 0; m < 2; ++m)
    for (int t = 0; t < 2; ++t)
      for (int r = 0; r < 8; ++r) { au[m][t].f[r] = 0.0f; av[m][t].f[r] = 0.0f; }

  const unsigned* abase0 = A + (size_t)(mg * 32 + lane15) * Kd2 + laneHi * 4;
  const unsigned* abase1 = abase0 + (size_t)16 * Kd2;
  const size_t boff = (size_t)(ng * 2) * Ktiles * 256 + lane * 8;

  for (int kt = 0; kt < Ktiles; ++kt) {
    FragAB a0, a1;
    a0.q[0] = *(const uint4*)(abase0 + kt * 16);
    a0.q[1] = *(const uint4*)(abase0 + kt * 16 + 8);
    a1.q[0] = *(const uint4*)(abase1 + kt * 16);
    a1.q[1] = *(const uint4*)(abase1 + kt * 16 + 8);
#pragma unroll
    for (int t = 0; t < 2; ++t) {
      const size_t to = boff + ((size_t)t * Ktiles + kt) * 256;
      FragAB b;
      b.q[0] = *(const uint4*)(B1 + to);
      b.q[1] = *(const uint4*)(B1 + to + 4);
      au[0][t].v = __builtin_amdgcn_wmma_f32_16x16x32_bf16(
          false, a0.v, false, b.v, (short)0, au[0][t].v, false, false);
      au[1][t].v = __builtin_amdgcn_wmma_f32_16x16x32_bf16(
          false, a1.v, false, b.v, (short)0, au[1][t].v, false, false);
      FragAB c;
      c.q[0] = *(const uint4*)(B3 + to);
      c.q[1] = *(const uint4*)(B3 + to + 4);
      av[0][t].v = __builtin_amdgcn_wmma_f32_16x16x32_bf16(
          false, a0.v, false, c.v, (short)0, av[0][t].v, false, false);
      av[1][t].v = __builtin_amdgcn_wmma_f32_16x16x32_bf16(
          false, a1.v, false, c.v, (short)0, av[1][t].v, false, false);
    }
  }

#pragma unroll
  for (int m = 0; m < 2; ++m) {
    const int row0 = mg * 32 + m * 16 + laneHi * 8;
#pragma unroll
    for (int t = 0; t < 2; ++t) {
      const int col = (ng * 2 + t) * 16 + lane15;
#pragma unroll
      for (int r = 0; r < 8; ++r) {
        float u = au[m][t].f[r], w = av[m][t].f[r];
        float s = u / (1.0f + expf(-u));  // silu
        out[(size_t)(row0 + r) * Nd + col] = f2bf(s * w);
      }
    }
  }
}

// -------------------------------------------------- gate input build + rmsnorm
__global__ void k_build_gi(const float* __restrict__ x, const float* __restrict__ ctx,
                           const int* __restrict__ positions,
                           const float* __restrict__ pos_table,
                           const float* __restrict__ norm_w,
                           float* __restrict__ gi_f, unsigned short* __restrict__ gi_b) {
  const int n = blockIdx.x, t = threadIdx.x;  // 256 threads per token
  __shared__ float row[DG];
  __shared__ float red[256];
  for (int d = t; d < H; d += 256) row[d] = x[(size_t)n * H + d];
  for (int d = t; d < CTXD; d += 256) row[H + d] = ctx[(size_t)n * CTXD + d];
  const int p = positions[n];
  for (int d = t; d < POSD; d += 256) row[H + CTXD + d] = pos_table[(size_t)p * POSD + d];
  __syncthreads();
  float s = 0.f;
  for (int d = t; d < DG; d += 256) { float v = row[d]; s += v * v; }
  red[t] = s; __syncthreads();
  for (int o = 128; o > 0; o >>= 1) { if (t < o) red[t] += red[t + o]; __syncthreads(); }
  const float rms = rsqrtf(red[0] / (float)DG + 1e-6f);
  for (int d = t; d < DG; d += 256) {
    float g = row[d] * rms * norm_w[d];
    gi_f[(size_t)n * DG + d] = g;
    gi_b[(size_t)n * DG + d] = f2bf(g);
  }
}

// -------------------------------------------- logits + softmax + top-2 (wave/token)
__global__ void k_router_head(const float* __restrict__ h, const float* __restrict__ proj,
                              const float* __restrict__ tempp,
                              float* __restrict__ o_idx, float* __restrict__ o_scr,
                              int* __restrict__ ew_e, float* __restrict__ ew_w,
                              float* __restrict__ scores) {
  const int lane = threadIdx.x & 31;
  const int n = blockIdx.x * (blockDim.x >> 5) + (threadIdx.x >> 5);
  float acc[NE];
#pragma unroll
  for (int e = 0; e < NE; ++e) acc[e] = 0.f;
  const float* hr = h + (size_t)n * DG;
  for (int k = lane; k < DG; k += 32) {
    const float hv = hr[k];
    const float* pw = proj + (size_t)k * NE;
#pragma unroll
    for (int e = 0; e < NE; ++e) acc[e] += hv * pw[e];
  }
#pragma unroll
  for (int e = 0; e < NE; ++e)
    for (int off = 16; off > 0; off >>= 1) acc[e] += __shfl_xor(acc[e], off);
  if (lane == 0) {
    const float temp = fmaxf(tempp[0], 0.3f);
    float mx = acc[0];
#pragma unroll
    for (int e = 1; e < NE; ++e) mx = fmaxf(mx, acc[e]);
    float p[NE], sum = 0.f;
#pragma unroll
    for (int e = 0; e < NE; ++e) { p[e] = expf((acc[e] - mx) / temp); sum += p[e]; }
    const float inv = 1.f / sum;
    float b1 = -1.f, b2 = -1.f; int i1 = 0, i2 = 0;
#pragma unroll
    for (int e = 0; e < NE; ++e) {
      const float v = p[e] * inv;
      scores[(size_t)n * NE + e] = v;
      if (v > b1)      { b2 = b1; i2 = i1; b1 = v; i1 = e; }
      else if (v > b2) { b2 = v;  i2 = e; }
    }
    o_idx[2 * n] = (float)i1; o_idx[2 * n + 1] = (float)i2;
    o_scr[2 * n] = b1;        o_scr[2 * n + 1] = b2;
    ew_e[2 * n]  = i1;        ew_e[2 * n + 1]  = i2;
    ew_w[2 * n]  = b1;        ew_w[2 * n + 1]  = b2;
  }
}

__global__ void k_reduce_me(const float* __restrict__ scores, float* __restrict__ me_sum) {
  const int e = blockIdx.x, t = threadIdx.x;
  __shared__ float red[256];
  float s = 0.f;
  for (int n = t; n < N_TOK; n += 256) s += scores[(size_t)n * NE + e];
  red[t] = s; __syncthreads();
  for (int o = 128; o > 0; o >>= 1) { if (t < o) red[t] += red[t + o]; __syncthreads(); }
  if (t == 0) me_sum[e] = red[0];
}

__global__ void k_zero_counts(int* ecount) { if (threadIdx.x < NE) ecount[threadIdx.x] = 0; }

// Greedy capacity dispatch == stable argsort(-w): rank = #{j' same-expert with
// (w' > w) || (w' == w && j' < j)}; assigned iff rank < CAP.
__global__ void k_dispatch(const int* __restrict__ ew_e, const float* __restrict__ ew_w,
                           float* __restrict__ o_mask, float* __restrict__ o_pos,
                           int* __restrict__ flags, int* __restrict__ ecount) {
  const int j = blockIdx.x * 256 + threadIdx.x;
  const int e = ew_e[j];
  const float w = ew_w[j];
  __shared__ int se[256];
  __shared__ float sw[256];
  int rank = 0;
  for (int c = 0; c < NK; c += 256) {
    se[threadIdx.x] = ew_e[c + threadIdx.x];
    sw[threadIdx.x] = ew_w[c + threadIdx.x];
    __syncthreads();
    for (int u = 0; u < 256; ++u) {
      const int jp = c + u;
      if (se[u] == e && (sw[u] > w || (sw[u] == w && jp < j))) rank++;
    }
    __syncthreads();
  }
  const bool ok = rank < CAP;
  o_mask[j] = ok ? 1.f : 0.f;
  o_pos[j] = ok ? (float)rank : 0.f;
  flags[j] = ok ? 1 : 0;
  if (ok) atomicAdd(&ecount[e], 1);
}

__global__ void k_overflow(const int* __restrict__ flags, float* __restrict__ o_ovf) {
  const int n = blockIdx.x * 256 + threadIdx.x;
  if (n < N_TOK) o_ovf[n] = (flags[2 * n] | flags[2 * n + 1]) ? 0.f : 1.f;
}

__global__ void k_next_ctx(const float* __restrict__ ctx, const int* __restrict__ positions,
                           const float* __restrict__ pos_table, float* __restrict__ o_nctx) {
  const int b = blockIdx.x, d = threadIdx.x;
  if (d >= CTXD + POSD) return;
  float s = 0.f;
  if (d < CTXD) {
    for (int t = 0; t < 2048; ++t) s += ctx[((size_t)b * 2048 + t) * CTXD + d];
  } else {
    const int dd = d - CTXD;
    for (int t = 0; t < 2048; ++t)
      s += pos_table[(size_t)positions[b * 2048 + t] * POSD + dd];
  }
  o_nctx[b * (CTXD + POSD) + d] = s * (1.f / 2048.f);
}

__global__ void k_finalize(const float* __restrict__ me_sum, const int* __restrict__ ecount,
                           float* __restrict__ o_cnt, float* __restrict__ o_aux) {
  const int t = threadIdx.x;
  if (t < NE) o_cnt[t] = (float)ecount[t];
  if (t == 0) {
    float aux = 0.f;
    for (int e = 0; e < NE; ++e)
      aux += (me_sum[e] / (float)N_TOK) * ((float)ecount[e] / (float)N_TOK);
    o_aux[0] = 0.01f * (float)NE * aux;
  }
}

// ------------------------------------------------------------------- host
extern "C" void kernel_launch(void* const* d_in, const int* in_sizes, int n_in,
                              void* d_out, int out_size, void* d_ws, size_t ws_size,
                              hipStream_t stream) {
  const float* x        = (const float*)d_in[0];
  const int*   positions= (const int*)  d_in[1];
  const float* w_c1     = (const float*)d_in[2];
  const float* b_c1     = (const float*)d_in[3];
  const float* w_c2     = (const float*)d_in[4];
  const float* b_c2     = (const float*)d_in[5];
  const float* norm_w   = (const float*)d_in[6];
  const float* mlp_w1   = (const float*)d_in[7];
  const float* mlp_w3   = (const float*)d_in[8];
  const float* mlp_w2   = (const float*)d_in[9];
  const float* proj_w   = (const float*)d_in[10];
  const float* pos_tab  = (const float*)d_in[11];
  const float* temp     = (const float*)d_in[12];

  char* wp = (char*)d_ws;
  auto carve = [&](size_t bytes) -> void* {
    void* p = (void*)wp;
    wp += (bytes + 255) & ~(size_t)255;
    return p;
  };
  unsigned short* xbf  = (unsigned short*)carve((size_t)N_TOK * H * 2);
  unsigned* wc1p = (unsigned*)carve((size_t)H * H2 * 2);
  unsigned* wc2p = (unsigned*)carve((size_t)H2 * CTXD * 2);
  unsigned* w1p  = (unsigned*)carve((size_t)DG * HID * 2);
  unsigned* w3p  = (unsigned*)carve((size_t)DG * HID * 2);
  unsigned* w2p  = (unsigned*)carve((size_t)HID * DG * 2);
  unsigned short* cmid = (unsigned short*)carve((size_t)N_TOK * H2 * 2);
  float* ctxf    = (float*)carve((size_t)N_TOK * CTXD * 4);
  float* gif     = (float*)carve((size_t)N_TOK * DG * 4);
  unsigned short* gib  = (unsigned short*)carve((size_t)N_TOK * DG * 2);
  unsigned short* hmid = (unsigned short*)carve((size_t)N_TOK * HID * 2);
  float* hbuf    = (float*)carve((size_t)N_TOK * DG * 4);
  float* scores  = (float*)carve((size_t)N_TOK * NE * 4);
  int*   ew_e    = (int*)  carve((size_t)NK * 4);
  float* ew_w    = (float*)carve((size_t)NK * 4);
  int*   flags   = (int*)  carve((size_t)NK * 4);
  float* me_sum  = (float*)carve(NE * 4);
  int*   ecount  = (int*)  carve(NE * 4);

  float* out    = (float*)d_out;
  float* o_idx  = out;
  float* o_scr  = out + 16384;
  float* o_mask = out + 32768;
  float* o_pos  = out + 49152;
  float* o_ovf  = out + 65536;
  float* o_cnt  = out + 73728;
  float* o_aux  = out + 73736;
  float* o_nctx = out + 73737;

  // activations -> bf16
  k_f32_to_bf16<<<(N_TOK * H + 255) / 256, 256, 0, stream>>>(x, xbf, N_TOK * H);
  // weights -> WMMA-B packed bf16
  k_pack_b<<<(H * H2 / 2 + 255) / 256, 256, 0, stream>>>(w_c1, wc1p, H2, H / 32);
  k_pack_b<<<(H2 * CTXD / 2 + 255) / 256, 256, 0, stream>>>(w_c2, wc2p, CTXD, H2 / 32);
  k_pack_b<<<(DG * HID / 2 + 255) / 256, 256, 0, stream>>>(mlp_w1, w1p, HID, DG / 32);
  k_pack_b<<<(DG * HID / 2 + 255) / 256, 256, 0, stream>>>(mlp_w3, w3p, HID, DG / 32);
  k_pack_b<<<(HID * DG / 2 + 255) / 256, 256, 0, stream>>>(mlp_w2, w2p, DG, HID / 32);

  // G1: cmid = gelu(x @ w_c1 + b_c1)   [8192 x 2048] bf16   (MT=2, NT=4)
  k_gemm_wmma<4, 0><<<(N_TOK / 32) * (H2 / 64) / 4, 128, 0, stream>>>(
      (const unsigned*)xbf, wc1p, cmid, b_c1, H2 / 64, H2, H / 32, H / 2);
  // G2: ctx = cmid @ w_c2 + b_c2       [8192 x 64] f32      (MT=2, NT=4)
  k_gemm_wmma<4, 1><<<(N_TOK / 32) * (CTXD / 64) / 4, 128, 0, stream>>>(
      (const unsigned*)cmid, wc2p, ctxf, b_c2, CTXD / 64, CTXD, H2 / 32, H2 / 2);
  // gi = rmsnorm(concat(x, ctx, pos_emb))
  k_build_gi<<<N_TOK, 256, 0, stream>>>(x, ctxf, positions, pos_tab, norm_w, gif, gib);
  // G3: hmid = silu(gi@w1) * (gi@w3)   [8192 x 3072] bf16   (MT=2, NT=2 dual)
  k_gemm_swiglu<<<(N_TOK / 32) * (HID / 32) / 4, 128, 0, stream>>>(
      (const unsigned*)gib, w1p, w3p, hmid, HID / 32, HID, DG / 32, DG / 2);
  // G4: h = hmid @ w2 + gi             [8192 x 1120] f32    (MT=2, NT=2)
  k_gemm_wmma<2, 2><<<(N_TOK / 32) * (DG / 32) / 4, 128, 0, stream>>>(
      (const unsigned*)hmid, w2p, hbuf, gif, DG / 32, DG, HID / 32, HID / 2);

  // head: logits, softmax, top-2
  k_router_head<<<N_TOK / 8, 256, 0, stream>>>(hbuf, proj_w, temp, o_idx, o_scr,
                                               ew_e, ew_w, scores);
  k_reduce_me<<<NE, 256, 0, stream>>>(scores, me_sum);
  k_zero_counts<<<1, 32, 0, stream>>>(ecount);
  k_dispatch<<<NK / 256, 256, 0, stream>>>(ew_e, ew_w, o_mask, o_pos, flags, ecount);
  k_overflow<<<(N_TOK + 255) / 256, 256, 0, stream>>>(flags, o_ovf);
  k_next_ctx<<<4, 128, 0, stream>>>(ctxf, positions, pos_tab, o_nctx);
  k_finalize<<<1, 32, 0, stream>>>(me_sum, ecount, o_cnt, o_aux);
}
